// MultiheadAttention_83167746719862
// MI455X (gfx1250) — compile-verified, matrix operand order checked by reading, not run on hardware
//
#include <hip/hip_runtime.h>
#include <hip/hip_bf16.h>
#include <stdint.h>

// ---- vector types matching CDNA5 WMMA builtin signatures ----
typedef __bf16        v16bf __attribute__((ext_vector_type(16)));
typedef float         v8f   __attribute__((ext_vector_type(8)));
typedef unsigned int  v4ui  __attribute__((ext_vector_type(4)));
typedef int           v8i   __attribute__((ext_vector_type(8)));
typedef int           v4i   __attribute__((ext_vector_type(4)));

union Frag16 { unsigned int u[8]; v16bf v; };

// round-to-nearest-even f32->bf16 (epilogue stores only)
__device__ __forceinline__ unsigned short f32_bf16(float f) {
  unsigned int u = __float_as_uint(f);
  u += 0x7FFFu + ((u >> 16) & 1u);
  return (unsigned short)(u >> 16);
}
// single-instruction truncating pack: v_perm_b32 -> {hi[31:16], lo[31:16]}
__device__ __forceinline__ unsigned int pack_bf16_trunc(float lo, float hi) {
  return __builtin_amdgcn_perm(__float_as_uint(hi), __float_as_uint(lo),
                               0x07060302u);
}

// ============================================================
// Kernel 1: qkv = x0 @ W^T -> Q (pre-scaled 1/D), K, V as bf16.
// M=8192, N=384, K=2048; one 16x16 tile per wave, bf16 WMMA.
// x0 (64MB) + W (3MB) are L2-resident (192MB L2) so re-reads are cheap.
// ============================================================
__global__ __launch_bounds__(256) void qkv_proj_kernel(
    const float* __restrict__ x0, const float* __restrict__ W,
    unsigned short* __restrict__ Qo, unsigned short* __restrict__ Ko,
    unsigned short* __restrict__ Vo) {
  constexpr int HC = 2048, D = 128;
  const int lane = threadIdx.x & 31;
  const int wave = threadIdx.x >> 5;
  const int gw   = blockIdx.x * 8 + wave;   // 0..12287
  const int mt   = gw & 511;                // 512 M-tiles
  const int nt   = gw >> 9;                 // 24 N-tiles
  const int m0 = mt * 16, n0 = nt * 16;
  const int r = lane & 15, half = lane >> 4;

  const float* xrow = x0 + (size_t)(m0 + r) * HC + (half << 3);  // A: lane = row M
  const float* wrow = W  + (size_t)(n0 + r) * HC + (half << 3);  // B: lane = col N

  v8f acc = {};
  for (int kc = 0; kc < HC; kc += 32) {
    Frag16 a, b;
    // vv 0..3 cover K = half*8..+7 (contiguous), vv 4..7 cover K = 16+half*8..+7
#pragma unroll
    for (int g = 0; g < 2; ++g) {          // g=0 -> K base 0, g=1 -> K base 16
      const float4 fa0 = *(const float4*)(xrow + kc + g * 16);
      const float4 fa1 = *(const float4*)(xrow + kc + g * 16 + 4);
      const float4 fb0 = *(const float4*)(wrow + kc + g * 16);
      const float4 fb1 = *(const float4*)(wrow + kc + g * 16 + 4);
      a.u[g * 4 + 0] = pack_bf16_trunc(fa0.x, fa0.y);
      a.u[g * 4 + 1] = pack_bf16_trunc(fa0.z, fa0.w);
      a.u[g * 4 + 2] = pack_bf16_trunc(fa1.x, fa1.y);
      a.u[g * 4 + 3] = pack_bf16_trunc(fa1.z, fa1.w);
      b.u[g * 4 + 0] = pack_bf16_trunc(fb0.x, fb0.y);
      b.u[g * 4 + 1] = pack_bf16_trunc(fb0.z, fb0.w);
      b.u[g * 4 + 2] = pack_bf16_trunc(fb1.x, fb1.y);
      b.u[g * 4 + 3] = pack_bf16_trunc(fb1.z, fb1.w);
    }
    acc = __builtin_amdgcn_wmma_f32_16x16x32_bf16(false, a.v, false, b.v,
                                                  (short)0, acc, false, false);
  }

  // Region depends only on nt (wave-uniform): pick base/col/scale once,
  // then a straight-line store loop (no per-element exec-mask churn).
  unsigned short* dstbase;
  int coloff;
  float scale;
  if (nt < 8)        { dstbase = Qo; coloff = n0;       scale = 1.0f / 128.0f; }
  else if (nt < 16)  { dstbase = Ko; coloff = n0 - 128; scale = 1.0f; }
  else               { dstbase = Vo; coloff = n0 - 256; scale = 1.0f; }
  unsigned short* dst = dstbase + (size_t)(m0 + (half << 3)) * D + coloff + r;
#pragma unroll
  for (int vv = 0; vv < 8; ++vv)           // C layout: M = vv + 8*half
    dst[(size_t)vv * D] = f32_bf16(acc[vv] * scale);
}

// ============================================================
// TDM: 1-D tensor_load_to_lds of contiguous bf16 (D# per ISA 08).
// LDS padding: +1 DWORD after every 64 DWORDs (one 128-elem bf16 row)
// -> 65-dword row stride in LDS, bank-conflict-free WMMA fragment reads.
// 6-arg builtin: (v4ui, v8i, v4i, v4i, v8i, cpol)
// ============================================================
__device__ __forceinline__ void tdm_load_k_tile(unsigned lds_off_bytes,
                                                const unsigned short* gsrc,
                                                unsigned nelem) {
  const unsigned long long ga = (unsigned long long)(uintptr_t)gsrc;
  v4ui g0;
  g0[0] = 1u;                                        // count=1, user descriptor
  g0[1] = lds_off_bytes;                             // lds_addr (bytes)
  g0[2] = (unsigned)ga;                              // global_addr[31:0]
  g0[3] = ((unsigned)(ga >> 32) & 0x01FFFFFFu)       // global_addr[56:32]
          | (2u << 30);                              // type = 2 (image)
  v8i g1;
  g1[0] = (int)((1u << 16)                           // data_size=1 (2 bytes)
                | (1u << 20)                         // pad_enable
                | (5u << 22));                       // pad_interval: 64 DWORDs
                                                     // pad_amount[31:25]=0: 1 DWORD
  g1[1] = (int)((nelem & 0xFFFFu) << 16);            // tensor_dim0[15:0]
  g1[2] = (int)(((nelem >> 16) & 0xFFFFu)            // tensor_dim0[31:16]
                | (1u << 16));                       // tensor_dim1 = 1
  g1[3] = (int)((nelem & 0xFFFFu) << 16);            // tile_dim0 = nelem
  g1[4] = 0;                                         // tile_dim1/2 = 0 (1-D)
  g1[5] = (int)nelem;                                // tensor_dim0_stride
  g1[6] = 0;
  g1[7] = 0;
  const v4i z4 = {0, 0, 0, 0};                       // groups 2/3 unused (<=2D)
  const v8i z8 = {0, 0, 0, 0, 0, 0, 0, 0};
  __builtin_amdgcn_tensor_load_to_lds(g0, g1, z4, z4, z8, 0);
}

// ============================================================
// Kernel 2: flash attention. 256 blocks x 8 waves; BM=32 Q-rows/block,
// BN=64 KV-cols/tile, d=128. bf16 WMMA, f32 online softmax in LDS.
// K: TDM double-buffered (padded rows). V: staged transposed (k-contiguous).
// Q: A-fragments pinned in registers. ~58.5 KB LDS.
// ============================================================
__global__ __launch_bounds__(256) void flash_attn_kernel(
    const unsigned short* __restrict__ Qg,
    const unsigned short* __restrict__ Kg,
    const unsigned short* __restrict__ Vg,
    const unsigned char* __restrict__ mask,
    float* __restrict__ out) {
  constexpr int N = 8192, D = 128, BM = 32, BN = 64, NT = N / BN;
  constexpr int KSTR = 130;  // padded K row stride (ushorts) = 65 dwords
  constexpr int VSTR = 66;   // padded Vt row stride (ushorts)
  constexpr int SSTR = 66;   // padded S row stride (floats)
  __shared__ __align__(16) unsigned short Ks[2][BN * KSTR];  // 2 x 16.6 KB
  __shared__ __align__(16) unsigned short Vt[D * VSTR];      // 16.5 KB, [n][k]
  __shared__ __align__(16) float Ss[BM * SSTR];              // 8.25 KB (S, then P)
  __shared__ float m_s[BM], l_s[BM], c_s[BM];

  const int tid  = threadIdx.x;
  const int lane = tid & 31, wave = tid >> 5;
  const int r = lane & 15, half = lane >> 4;
  const int row0 = blockIdx.x * BM;
  const int sr0 = (wave & 1) * 16;  // this wave's O row block
  const int oc0 = (wave >> 1) * 32; // this wave's O col block
  const int sti = wave >> 2;        // this wave's S tile (2x4 of 16x16)
  const int stj = wave & 3;

  // ---- Q A-fragments in registers (Q pre-scaled by 1/D, bf16) ----
  Frag16 qa[4];
  {
    const unsigned short* qrow = Qg + (size_t)(row0 + sti * 16 + r) * D;
#pragma unroll
    for (int c = 0; c < 4; ++c)
#pragma unroll
      for (int vv = 0; vv < 8; ++vv) {
        const int kbv = ((vv & 4) << 2) + (half << 3) + ((vv & 3) << 1);
        qa[c].u[vv] = *(const unsigned int*)(qrow + c * 32 + kbv);
      }
  }
  if (tid < BM) { m_s[tid] = -1e30f; l_s[tid] = 0.0f; }
  v8f acc[2] = {};

  // prologue: kick off first K tile DMA
  if (wave == 0)
    tdm_load_k_tile((unsigned)(uintptr_t)(void*)&Ks[0][0], Kg, BN * D);

  for (int kb = 0; kb < NT; ++kb) {
    const unsigned short* kbuf = &Ks[kb & 1][0];
    // ---- pipeline: issue K(kb+1), wait only for K(kb) (in-order TDM) ----
    if (wave == 0) {
      if (kb + 1 < NT) {
        tdm_load_k_tile((unsigned)(uintptr_t)(void*)&Ks[(kb + 1) & 1][0],
                        Kg + (size_t)(kb + 1) * BN * D, BN * D);
        __builtin_amdgcn_s_wait_tensorcnt(1);
      } else {
        __builtin_amdgcn_s_wait_tensorcnt(0);
      }
    }
    // ---- stage V transposed: Vt[n][k] so P@V B-fragments are k-contiguous ----
    {
      const unsigned int* vsrc = (const unsigned int*)(Vg + (size_t)kb * BN * D);
#pragma unroll
      for (int i = 0; i < (BN * D / 2) / 256; ++i) {   // 16 dwords/thread
        const int idx = tid + i * 256;
        const unsigned int w = vsrc[idx];
        const int k  = idx >> 6;             // V row (64 dwords per row)
        const int n2 = (idx & 63) << 1;      // V col pair
        Vt[n2 * VSTR + k]       = (unsigned short)(w & 0xFFFFu);
        Vt[(n2 + 1) * VSTR + k] = (unsigned short)(w >> 16);
      }
    }
    // ---- prefetch next mask tile (the only real HBM stream) ----
    if (tid < BM) {
      const int pkb = (kb + 1 < NT) ? kb + 1 : kb;
      __builtin_prefetch(mask + (size_t)(row0 + tid) * N + (size_t)pkb * BN, 0, 0);
    }
    __syncthreads();

    // ---- S = Q K^T (one 16x16 tile per wave) ----
    {
      v8f s = {};
#pragma unroll
      for (int c = 0; c < 4; ++c) {
        Frag16 b;
#pragma unroll
        for (int vv = 0; vv < 8; ++vv) {
          const int kbv = ((vv & 4) << 2) + (half << 3) + ((vv & 3) << 1);
          b.u[vv] = *(const unsigned int*)&kbuf[(stj * 16 + r) * KSTR + c * 32 + kbv];
        }
        s = __builtin_amdgcn_wmma_f32_16x16x32_bf16(false, qa[c].v, false, b.v,
                                                    (short)0, s, false, false);
      }
      // apply boolean mask, spill S tile to LDS
#pragma unroll
      for (int vv = 0; vv < 8; ++vv) {
        const int lr = sti * 16 + vv + (half << 3);
        const int lc = stj * 16 + r;
        const size_t gi = (size_t)(row0 + lr) * N + (size_t)(kb * BN + lc);
        Ss[lr * SSTR + lc] = mask[gi] ? -1e30f : s[vv];
      }
    }
    __syncthreads();

    // ---- online softmax row stats; P = exp(S - m_new) overwrites S ----
    if (tid < BM) {
      float* srow = &Ss[tid * SSTR];
      const float mold = m_s[tid];
      float mmax = mold;
#pragma unroll
      for (int j = 0; j < BN; j += 2) {
        const float2 v2 = *(const float2*)(srow + j);
        mmax = fmaxf(mmax, fmaxf(v2.x, v2.y));
      }
      const float corr = __expf(mold - mmax);
      float sum = 0.0f;
#pragma unroll
      for (int j = 0; j < BN; j += 2) {
        float2 v2 = *(const float2*)(srow + j);
        v2.x = __expf(v2.x - mmax);
        v2.y = __expf(v2.y - mmax);
        *(float2*)(srow + j) = v2;
        sum += v2.x + v2.y;
      }
      m_s[tid] = mmax;
      l_s[tid] = l_s[tid] * corr + sum;
      c_s[tid] = corr;
    }
    __syncthreads();

    // ---- O = O*corr + P @ V ----
    float scl[8];
#pragma unroll
    for (int vv = 0; vv < 8; ++vv) scl[vv] = c_s[sr0 + vv + (half << 3)];
#pragma unroll
    for (int t = 0; t < 2; ++t)
#pragma unroll
      for (int vv = 0; vv < 8; ++vv) acc[t][vv] *= scl[vv];

#pragma unroll
    for (int kk = 0; kk < BN; kk += 32) {
      Frag16 a;  // A = P rows sr0..sr0+15, K-chunk kk..kk+31 (f32 -> bf16)
#pragma unroll
      for (int vv = 0; vv < 8; ++vv) {
        const int kbv = ((vv & 4) << 2) + (half << 3) + ((vv & 3) << 1);
        a.u[vv] = pack_bf16_trunc(Ss[(sr0 + r) * SSTR + kk + kbv],
                                  Ss[(sr0 + r) * SSTR + kk + kbv + 1]);
      }
#pragma unroll
      for (int t = 0; t < 2; ++t) {
        Frag16 b;  // B from transposed V: k-contiguous dword loads
        const int col = oc0 + t * 16 + r;
#pragma unroll
        for (int vv = 0; vv < 8; ++vv) {
          const int kbv = ((vv & 4) << 2) + (half << 3) + ((vv & 3) << 1);
          b.u[vv] = *(const unsigned int*)&Vt[col * VSTR + kk + kbv];
        }
        acc[t] = __builtin_amdgcn_wmma_f32_16x16x32_bf16(false, a.v, false, b.v,
                                                         (short)0, acc[t], false, false);
      }
    }
    __syncthreads();   // protect K/Vt/Ss before next iteration's overwrite
  }

  // ---- epilogue: O / l ----
#pragma unroll
  for (int vv = 0; vv < 8; ++vv) {
    const int lr = sr0 + vv + (half << 3);
    const float inv = 1.0f / l_s[lr];
#pragma unroll
    for (int t = 0; t < 2; ++t) {
      const int col = oc0 + t * 16 + r;
      out[(size_t)(row0 + lr) * D + col] = acc[t][vv] * inv;
    }
  }
}

// ============================================================
extern "C" void kernel_launch(void* const* d_in, const int* in_sizes, int n_in,
                              void* d_out, int out_size, void* d_ws, size_t ws_size,
                              hipStream_t stream) {
  (void)in_sizes; (void)n_in; (void)out_size; (void)ws_size;
  const float*         x0   = (const float*)d_in[0];
  const unsigned char* mask = (const unsigned char*)d_in[1];  // numpy bool, 1B
  const float*         W    = (const float*)d_in[2];
  float* out = (float*)d_out;

  // workspace: Q | K | V, each 8192x128 bf16 (2 MB each, 6 MB total)
  unsigned short* Qw = (unsigned short*)d_ws;
  unsigned short* Kw = Qw + (size_t)8192 * 128;
  unsigned short* Vw = Kw + (size_t)8192 * 128;

  // 12288 output tiles (512 x 24) -> 8 waves/block -> 1536 blocks
  qkv_proj_kernel<<<1536, 256, 0, stream>>>(x0, W, Qw, Kw, Vw);
  // 8192 / BM(32) = 256 blocks
  flash_attn_kernel<<<256, 256, 0, stream>>>(Qw, Kw, Vw, mask, out);
}
